// AttentionPE_11020886082043
// MI455X (gfx1250) — matrix operand order repaired and checked by reading, not compiled, not run on hardware
//
#include <hip/hip_runtime.h>
#include <hip/hip_bf16.h>

// ---------------------------------------------------------------------------
// AttentionPE for MI455X (gfx1250): bf16 WMMA GEMMs for all convs, wave32
// shuffle attention. D = A*B + C via v_wmma_f32_16x16x32_bf16.
// LDS tiles are stored [pixel][channel] so every WMMA B fragment is a single
// contiguous 32B LDS read (2x ds_load_b128); dilated weights are pre-
// transposed to [tap][O][I] so A fragments are contiguous global_load_b128.
// ---------------------------------------------------------------------------

typedef __attribute__((ext_vector_type(16))) __bf16 v16bf;
typedef __attribute__((ext_vector_type(8)))  float  v8f;

#define C_    64
#define H_    512
#define W_    512
#define HW_   ((size_t)H_ * (size_t)W_)
#define ACCEL 8
#define WC_   (W_ / ACCEL)   /* 64 : column spacing inside an attention block */
#define LDSP  72             /* padded channel pitch of LDS tiles (16B-aligned rows) */

// ---------------------------------------------------------------------------
// Positional encoding: pe[h][w] = sin(h * 10000^(-(2*(w/2))/W))
// ---------------------------------------------------------------------------
__device__ __forceinline__ float pos_enc(int h, int w) {
  float e = -(2.0f * (float)(w >> 1)) / (float)W_;
  float f = __expf(e * 9.210340371976184f);  // ln(10000)
  return __sinf((float)h * f);
}

// ---------------------------------------------------------------------------
// A fragment (16x32 bf16) from a row-major [64 x 64] bf16 weight matrix.
// Lane half 0 -> K {0..15}, half 1 -> K {16..31} (matches contiguous B).
// K indices are contiguous (16) -> compiler emits 2x global b128.
// ---------------------------------------------------------------------------
__device__ __forceinline__ v16bf load_a_frag(const __bf16* __restrict__ wmat,
                                             int mt, int lane, int s) {
  const int lm = lane & 15, lh = lane >> 4;
  const int m = mt * 16 + lm;
  const int kbase = 32 * s + lh * 16;
  v16bf a;
#pragma unroll
  for (int j = 0; j < 16; ++j) {
    const int v = j >> 1;
    const int koff = 2 * v;
    a[j] = wmat[m * 64 + kbase + koff + (j & 1)];
  }
  return a;
}

// ---------------------------------------------------------------------------
// K0a: fp32 -> bf16 weight conversion (1x1 conv weights)
// ---------------------------------------------------------------------------
__global__ void cvt_bf16_kernel(const float* __restrict__ src,
                                __bf16* __restrict__ dst, int n) {
  int i = blockIdx.x * blockDim.x + threadIdx.x;
  if (i < n) dst[i] = (__bf16)src[i];
}

// ---------------------------------------------------------------------------
// K0b: dilated conv weights [O][I][3][3] fp32 -> [tap][O][I] bf16
// ---------------------------------------------------------------------------
__global__ void cvt_dilw_kernel(const float* __restrict__ src,
                                __bf16* __restrict__ dst) {
  int idx = blockIdx.x * blockDim.x + threadIdx.x;
  if (idx < C_ * C_ * 9) {
    const int o   = idx / (C_ * 9);
    const int r   = idx % (C_ * 9);
    const int i   = r / 9;
    const int tap = r % 9;
    dst[((size_t)tap * C_ + o) * C_ + i] = (__bf16)src[idx];
  }
}

// ---------------------------------------------------------------------------
// K1: instance-norm statistics (one workgroup per channel)
// stats[c] = mean, stats[64+c] = rsqrt(var + eps)
// ---------------------------------------------------------------------------
__global__ void __launch_bounds__(256)
stats_kernel(const float* __restrict__ x, float* __restrict__ stats) {
  __shared__ float ssum[256];
  __shared__ float ssq[256];
  const int c = blockIdx.x;
  const float* p = x + (size_t)c * HW_;
  float s = 0.f, q = 0.f;
  for (size_t i = threadIdx.x; i < HW_; i += 256) {
    float v = p[i];
    s += v;
    q += v * v;
  }
  ssum[threadIdx.x] = s;
  ssq[threadIdx.x] = q;
  __syncthreads();
  for (int d = 128; d > 0; d >>= 1) {
    if ((int)threadIdx.x < d) {
      ssum[threadIdx.x] += ssum[threadIdx.x + d];
      ssq[threadIdx.x]  += ssq[threadIdx.x + d];
    }
    __syncthreads();
  }
  if (threadIdx.x == 0) {
    float mu  = ssum[0] / (float)HW_;
    float var = ssq[0] / (float)HW_ - mu * mu;
    stats[c]      = mu;
    stats[64 + c] = rsqrtf(var + 1e-5f);
  }
}

// GEMM core shared by q/k/v (bf16 out). Each wave: one 16-row M tile,
// two 16-col N subtiles (cols n0 and n0+32), K=64 in two 32-steps.
#define WMMA_GEMM_STORE_BF16(wmat, bias, outp)                                 \
  {                                                                            \
    v8f acc0, acc1;                                                            \
    _Pragma("unroll") for (int v = 0; v < 8; ++v) {                            \
      const float b_ = (bias)[mt * 16 + lh * 8 + v];                           \
      acc0[v] = b_;                                                            \
      acc1[v] = b_;                                                            \
    }                                                                          \
    const v16bf a0 = load_a_frag((wmat), mt, lane, 0);                         \
    const v16bf a1 = load_a_frag((wmat), mt, lane, 1);                         \
    acc0 = __builtin_amdgcn_wmma_f32_16x16x32_bf16(false, a0, false, b00, (short)0, acc0, false, false); \
    acc0 = __builtin_amdgcn_wmma_f32_16x16x32_bf16(false, a1, false, b01, (short)0, acc0, false, false); \
    acc1 = __builtin_amdgcn_wmma_f32_16x16x32_bf16(false, a0, false, b10, (short)0, acc1, false, false); \
    acc1 = __builtin_amdgcn_wmma_f32_16x16x32_bf16(false, a1, false, b11, (short)0, acc1, false, false); \
    _Pragma("unroll") for (int v = 0; v < 8; ++v) {                            \
      const size_t rowoff = (size_t)(mt * 16 + lh * 8 + v) * HW_;              \
      (outp)[rowoff + col0] = (__bf16)acc0[v];                                 \
      (outp)[rowoff + col1] = (__bf16)acc1[v];                                 \
    }                                                                          \
  }

// ---------------------------------------------------------------------------
// K2: fused instance-norm + pos-enc + q/k/v 1x1 conv (bf16 WMMA GEMM)
// Workgroup = 256 thr (8 waves) computes M=64 x N=64 pixels, all 3 outputs.
// ---------------------------------------------------------------------------
__global__ void __launch_bounds__(256)
qkv_gemm_kernel(const float* __restrict__ x, const float* __restrict__ stats,
                const __bf16* __restrict__ wq, const __bf16* __restrict__ wk,
                const __bf16* __restrict__ wv,
                const float* __restrict__ bq, const float* __restrict__ bk,
                const float* __restrict__ bv,
                __bf16* __restrict__ oq, __bf16* __restrict__ ok,
                __bf16* __restrict__ ov) {
  __shared__ __bf16 lB[64][LDSP];  // [pixel][channel]
  const size_t p0 = (size_t)blockIdx.x * 64;
  const int row = (int)(p0 >> 9);
  const int w0  = (int)(p0 & (W_ - 1));

  for (int idx = threadIdx.x; idx < 64 * 64; idx += 256) {
    const int c = idx >> 6, n = idx & 63;           // coalesced global read
    const float mu = stats[c], rs = stats[64 + c];
    const float val = (x[(size_t)c * HW_ + p0 + n] - mu) * rs + pos_enc(row, w0 + n);
    lB[n][c] = (__bf16)val;
  }
  __syncthreads();

  const int lane = threadIdx.x & 31, wave = threadIdx.x >> 5;
  const int mt = wave & 3, nsub = wave >> 2;
  const int lm = lane & 15, lh = lane >> 4;
  const int n0 = nsub * 16 + lm;

  // B fragments: 16 contiguous K values per lane -> 2x ds_load_b128 each
  const __bf16* r0 = &lB[n0][lh * 16];
  const __bf16* r1 = &lB[n0 + 32][lh * 16];
  const v16bf b00 = *(const v16bf*)(r0);
  const v16bf b01 = *(const v16bf*)(r0 + 32);
  const v16bf b10 = *(const v16bf*)(r1);
  const v16bf b11 = *(const v16bf*)(r1 + 32);

  const size_t col0 = p0 + (size_t)n0;
  const size_t col1 = col0 + 32;

  WMMA_GEMM_STORE_BF16(wq, bq, oq);
  WMMA_GEMM_STORE_BF16(wk, bk, ok);
  WMMA_GEMM_STORE_BF16(wv, bv, ov);
}

// ---------------------------------------------------------------------------
// K3: dilated 3x3 conv (dilation 2, zero pad 2) as 9-tap WMMA accumulation.
// Weights pre-transposed to [tap][O][I].
// ---------------------------------------------------------------------------
__global__ void __launch_bounds__(256)
dil_gemm_kernel(const __bf16* __restrict__ in, const __bf16* __restrict__ wdt,
                const float* __restrict__ bias, __bf16* __restrict__ out) {
  __shared__ __bf16 lS[64][LDSP];  // [pixel][channel]
  const size_t p0 = (size_t)blockIdx.x * 64;
  const int row = (int)(p0 >> 9);
  const int w0  = (int)(p0 & (W_ - 1));

  const int lane = threadIdx.x & 31, wave = threadIdx.x >> 5;
  const int mt = wave & 3, nsub = wave >> 2;
  const int lm = lane & 15, lh = lane >> 4;
  const int n0 = nsub * 16 + lm;

  // Warm L2 for the next tile while we grind this one (global_prefetch_b8).
  {
    const int c = threadIdx.x >> 6, n = threadIdx.x & 63;
    __builtin_prefetch(in + (size_t)c * HW_ + p0 + 64 + n, 0, 1);
  }

  v8f acc0, acc1;
#pragma unroll
  for (int v = 0; v < 8; ++v) {
    const float b_ = bias[mt * 16 + lh * 8 + v];
    acc0[v] = b_;
    acc1[v] = b_;
  }

  for (int tap = 0; tap < 9; ++tap) {
    const int dy = 2 * ((tap / 3) - 1);
    const int dx = 2 * ((tap % 3) - 1);
    const int hr = row + dy;
    __syncthreads();  // previous tap's fragment reads complete before restage
    for (int idx = threadIdx.x; idx < 64 * 64; idx += 256) {
      const int c = idx >> 6, n = idx & 63;
      const int wcc = w0 + n + dx;
      __bf16 val = (__bf16)0.0f;
      if (hr >= 0 && hr < H_ && wcc >= 0 && wcc < W_)
        val = in[(size_t)c * HW_ + (size_t)hr * W_ + wcc];
      lS[n][c] = val;
    }
    __syncthreads();

    const __bf16* r0 = &lS[n0][lh * 16];
    const __bf16* r1 = &lS[n0 + 32][lh * 16];
    const v16bf b00 = *(const v16bf*)(r0);
    const v16bf b01 = *(const v16bf*)(r0 + 32);
    const v16bf b10 = *(const v16bf*)(r1);
    const v16bf b11 = *(const v16bf*)(r1 + 32);

    const __bf16* wt = wdt + (size_t)tap * C_ * C_;
    const v16bf a0 = load_a_frag(wt, mt, lane, 0);
    const v16bf a1 = load_a_frag(wt, mt, lane, 1);
    acc0 = __builtin_amdgcn_wmma_f32_16x16x32_bf16(false, a0, false, b00, (short)0, acc0, false, false);
    acc0 = __builtin_amdgcn_wmma_f32_16x16x32_bf16(false, a1, false, b01, (short)0, acc0, false, false);
    acc1 = __builtin_amdgcn_wmma_f32_16x16x32_bf16(false, a0, false, b10, (short)0, acc1, false, false);
    acc1 = __builtin_amdgcn_wmma_f32_16x16x32_bf16(false, a1, false, b11, (short)0, acc1, false, false);
  }

  const size_t col0 = p0 + (size_t)n0;
  const size_t col1 = col0 + 32;
#pragma unroll
  for (int v = 0; v < 8; ++v) {
    const size_t rowoff = (size_t)(mt * 16 + lh * 8 + v) * HW_;
    out[rowoff + col0] = (__bf16)acc0[v];
    out[rowoff + col1] = (__bf16)acc1[v];
  }
}

// ---------------------------------------------------------------------------
// K4: block attention. Block n = h*64 + wc holds 8 columns w = i*64 + wc.
// One wave per block; lane <-> (i in {l>>3, l>>3 + 4}, j = l&7).
// ---------------------------------------------------------------------------
__global__ void __launch_bounds__(256)
attn_kernel(const __bf16* __restrict__ q, const __bf16* __restrict__ k,
            const __bf16* __restrict__ v, __bf16* __restrict__ att) {
  const int lane = threadIdx.x & 31;
  const int wave = threadIdx.x >> 5;
  const int n  = blockIdx.x * 8 + wave;  // 0..32767
  const int h  = n >> 6;
  const int wc = n & 63;
  const int j  = lane & 7;
  const int i0 = lane >> 3;
  const int i1 = i0 + 4;
  const size_t rowbase = (size_t)h * W_ + (size_t)wc;

  float s0 = 0.f, s1 = 0.f;
  for (int c = 0; c < C_; ++c) {
    const size_t base = (size_t)c * HW_ + rowbase;
    const float kj = (float)k[base + (size_t)j * WC_];
    s0 += (float)q[base + (size_t)i0 * WC_] * kj;
    s1 += (float)q[base + (size_t)i1 * WC_] * kj;
  }
  s0 *= 0.125f;  // C^-0.5 = 1/8
  s1 *= 0.125f;

  float m0 = s0, m1 = s1;
#pragma unroll
  for (int d = 1; d < 8; d <<= 1) {
    m0 = fmaxf(m0, __shfl_xor(m0, d, 32));
    m1 = fmaxf(m1, __shfl_xor(m1, d, 32));
  }
  const float e0 = __expf(s0 - m0), e1 = __expf(s1 - m1);
  float t0 = e0, t1 = e1;
#pragma unroll
  for (int d = 1; d < 8; d <<= 1) {
    t0 += __shfl_xor(t0, d, 32);
    t1 += __shfl_xor(t1, d, 32);
  }
  const float pr0 = e0 / t0, pr1 = e1 / t1;

  for (int c = 0; c < C_; ++c) {
    const size_t base = (size_t)c * HW_ + rowbase;
    const float vj = (float)v[base + (size_t)j * WC_];
    float o0 = vj * pr0, o1 = vj * pr1;
#pragma unroll
    for (int d = 1; d < 8; d <<= 1) {
      o0 += __shfl_xor(o0, d, 32);
      o1 += __shfl_xor(o1, d, 32);
    }
    if (j == 0) {
      att[base + (size_t)i0 * WC_] = (__bf16)o0;
      att[base + (size_t)i1 * WC_] = (__bf16)o1;
    }
  }
}

// ---------------------------------------------------------------------------
// K5: proj 1x1 conv (WMMA) + residual add, fp32 output
// ---------------------------------------------------------------------------
__global__ void __launch_bounds__(256)
proj_gemm_kernel(const __bf16* __restrict__ att, const __bf16* __restrict__ wp,
                 const float* __restrict__ pb, const float* __restrict__ x,
                 float* __restrict__ out) {
  __shared__ __bf16 lB[64][LDSP];
  const size_t p0 = (size_t)blockIdx.x * 64;
  for (int idx = threadIdx.x; idx < 64 * 64; idx += 256) {
    const int c = idx >> 6, n = idx & 63;
    lB[n][c] = att[(size_t)c * HW_ + p0 + n];
  }
  __syncthreads();

  const int lane = threadIdx.x & 31, wave = threadIdx.x >> 5;
  const int mt = wave & 3, nsub = wave >> 2;
  const int lm = lane & 15, lh = lane >> 4;
  const int n0 = nsub * 16 + lm;

  const __bf16* r0 = &lB[n0][lh * 16];
  const __bf16* r1 = &lB[n0 + 32][lh * 16];
  const v16bf b00 = *(const v16bf*)(r0);
  const v16bf b01 = *(const v16bf*)(r0 + 32);
  const v16bf b10 = *(const v16bf*)(r1);
  const v16bf b11 = *(const v16bf*)(r1 + 32);

  v8f acc0, acc1;
#pragma unroll
  for (int v = 0; v < 8; ++v) {
    const float b_ = pb[mt * 16 + lh * 8 + v];
    acc0[v] = b_;
    acc1[v] = b_;
  }
  const v16bf a0 = load_a_frag(wp, mt, lane, 0);
  const v16bf a1 = load_a_frag(wp, mt, lane, 1);
  acc0 = __builtin_amdgcn_wmma_f32_16x16x32_bf16(false, a0, false, b00, (short)0, acc0, false, false);
  acc0 = __builtin_amdgcn_wmma_f32_16x16x32_bf16(false, a1, false, b01, (short)0, acc0, false, false);
  acc1 = __builtin_amdgcn_wmma_f32_16x16x32_bf16(false, a0, false, b10, (short)0, acc1, false, false);
  acc1 = __builtin_amdgcn_wmma_f32_16x16x32_bf16(false, a1, false, b11, (short)0, acc1, false, false);

  const size_t col0 = p0 + (size_t)n0;
  const size_t col1 = col0 + 32;
#pragma unroll
  for (int v = 0; v < 8; ++v) {
    const size_t rowoff = (size_t)(mt * 16 + lh * 8 + v) * HW_;
    out[rowoff + col0] = acc0[v] + x[rowoff + col0];
    out[rowoff + col1] = acc1[v] + x[rowoff + col1];
  }
}

// ---------------------------------------------------------------------------
// Host side
// ---------------------------------------------------------------------------
extern "C" void kernel_launch(void* const* d_in, const int* in_sizes, int n_in,
                              void* d_out, int out_size, void* d_ws, size_t ws_size,
                              hipStream_t stream) {
  const float* x      = (const float*)d_in[0];
  const float* q_w    = (const float*)d_in[1];
  const float* q_b    = (const float*)d_in[2];
  const float* k_w    = (const float*)d_in[3];
  const float* k_b    = (const float*)d_in[4];
  const float* v_w    = (const float*)d_in[5];
  const float* v_b    = (const float*)d_in[6];
  const float* dil_w  = (const float*)d_in[7];
  const float* dil_b  = (const float*)d_in[8];
  const float* proj_w = (const float*)d_in[9];
  const float* proj_b = (const float*)d_in[10];
  // d_in[11] = accel (compile-time 8)

  char* ws = (char*)d_ws;
  size_t off = 0;
  auto take = [&](size_t bytes) -> char* {
    char* p = ws + off;
    off += (bytes + 255) & ~(size_t)255;
    return p;
  };

  __bf16* wq  = (__bf16*)take((size_t)C_ * C_ * 2);
  __bf16* wk  = (__bf16*)take((size_t)C_ * C_ * 2);
  __bf16* wv  = (__bf16*)take((size_t)C_ * C_ * 2);
  __bf16* wdt = (__bf16*)take((size_t)C_ * C_ * 9 * 2);   // [tap][O][I]
  __bf16* wp  = (__bf16*)take((size_t)C_ * C_ * 2);
  float*  st  = (float*) take((size_t)2 * C_ * 4);
  __bf16* bq  = (__bf16*)take((size_t)C_ * HW_ * 2);
  __bf16* bk  = (__bf16*)take((size_t)C_ * HW_ * 2);
  __bf16* bv  = (__bf16*)take((size_t)C_ * HW_ * 2);
  __bf16* bqd = (__bf16*)take((size_t)C_ * HW_ * 2);
  __bf16* bkd = (__bf16*)take((size_t)C_ * HW_ * 2);
  __bf16* bvd = (__bf16*)take((size_t)C_ * HW_ * 2);
  __bf16* bat = (__bf16*)take((size_t)C_ * HW_ * 2);

  // weights -> bf16 (dilated weights transposed to [tap][O][I])
  cvt_bf16_kernel<<<(C_ * C_ + 255) / 256, 256, 0, stream>>>(q_w, wq, C_ * C_);
  cvt_bf16_kernel<<<(C_ * C_ + 255) / 256, 256, 0, stream>>>(k_w, wk, C_ * C_);
  cvt_bf16_kernel<<<(C_ * C_ + 255) / 256, 256, 0, stream>>>(v_w, wv, C_ * C_);
  cvt_dilw_kernel<<<(C_ * C_ * 9 + 255) / 256, 256, 0, stream>>>(dil_w, wdt);
  cvt_bf16_kernel<<<(C_ * C_ + 255) / 256, 256, 0, stream>>>(proj_w, wp, C_ * C_);

  // instance norm stats
  stats_kernel<<<C_, 256, 0, stream>>>(x, st);

  // fused norm + posenc + q/k/v 1x1 conv  (64-pixel tiles)
  const int ntiles = (int)(HW_ / 64);  // 4096
  qkv_gemm_kernel<<<ntiles, 256, 0, stream>>>(x, st, wq, wk, wv, q_b, k_b, v_b,
                                              bq, bk, bv);

  // dilated 3x3 convs
  dil_gemm_kernel<<<ntiles, 256, 0, stream>>>(bq, wdt, dil_b, bqd);
  dil_gemm_kernel<<<ntiles, 256, 0, stream>>>(bk, wdt, dil_b, bkd);
  dil_gemm_kernel<<<ntiles, 256, 0, stream>>>(bv, wdt, dil_b, bvd);

  // block attention (32768 blocks, 8 per workgroup)
  attn_kernel<<<32768 / 8, 256, 0, stream>>>(bqd, bkd, bvd, bat);

  // proj + residual
  proj_gemm_kernel<<<ntiles, 256, 0, stream>>>(bat, wp, proj_b, x, (float*)d_out);
}
